// GQAAttention_1889785610988
// MI455X (gfx1250) — matrix operand order repaired and checked by reading, not compiled, not run on hardware
//
#include <hip/hip_runtime.h>
#include <hip/hip_bf16.h>
#include <math.h>

// ---------------- types ----------------
typedef __bf16 bf16_t;
typedef __attribute__((ext_vector_type(16))) __bf16 v16bf;
typedef __attribute__((ext_vector_type(8)))  float  v8f;
typedef __attribute__((ext_vector_type(4)))  unsigned int v4u;
typedef __attribute__((ext_vector_type(4)))  int v4i;
typedef __attribute__((ext_vector_type(8)))  int v8i;

union Frag16 {            // 16 bf16 = 8 VGPRs = 32 bytes
    v16bf v;
    uint4 u[2];
};

__device__ __forceinline__ v8f wmma_bf16(v16bf a, v16bf b, v8f c) {
    return __builtin_amdgcn_wmma_f32_16x16x32_bf16(false, a, false, b,
                                                   (short)0, c, false, false);
}

// ---------------- problem constants ----------------
#define T_SEQ 2048
#define HID   4096
#define NQ    32
#define NKV   8
#define HD    128
#define GROUP (NQ / NKV)

// =====================================================================
// Tensor Data Mover: 2D bf16 tile DMA global -> LDS.
// D# packing per CDNA5 ISA.
// tensor dims == tile dims (problem sizes are exact multiples, no OOB).
// =====================================================================
__device__ __forceinline__ void tdm_load_2d(unsigned lds_off, const void* gptr,
                                            unsigned tile_d0, unsigned tile_d1,
                                            unsigned long long stride_elems) {
    unsigned long long ga = (unsigned long long)(uintptr_t)gptr;
    v4u g0;
    g0.x = 1u;                                                  // count=1 (valid), no gather
    g0.y = lds_off;                                             // lds_addr
    g0.z = (unsigned)ga;                                        // global_addr[31:0]
    g0.w = (unsigned)((ga >> 32) & 0x01FFFFFFu) | (2u << 30);   // addr[56:32] | type=2
    v8i g1;
    g1[0] = (int)(1u << 16);                                    // data_size=1 (2 bytes)
    g1[1] = (int)((tile_d0 & 0xFFFFu) << 16);                   // tensor_dim0 lo16 @ bits63:48
    g1[2] = (int)(((tile_d0 >> 16) & 0xFFFFu) |
                  ((tile_d1 & 0xFFFFu) << 16));                 // dim0 hi | tensor_dim1 lo
    g1[3] = (int)(((tile_d1 >> 16) & 0xFFFFu) |
                  ((tile_d0 & 0xFFFFu) << 16));                 // dim1 hi | tile_dim0
    g1[4] = (int)(tile_d1 & 0xFFFFu);                           // tile_dim1 ; tile_dim2=0
    g1[5] = (int)(unsigned)(stride_elems & 0xFFFFFFFFu);        // tensor_dim0_stride lo32
    g1[6] = (int)(unsigned)((stride_elems >> 32) & 0xFFFFu);    // stride hi16 | dim1_stride lo=0
    g1[7] = 0;
    v4i z4 = {0, 0, 0, 0};
#if defined(__clang_major__) && (__clang_major__ >= 23)
    v8i z8 = {0, 0, 0, 0, 0, 0, 0, 0};
    __builtin_amdgcn_tensor_load_to_lds(g0, g1, z4, z4, z8, 0);
#else
    __builtin_amdgcn_tensor_load_to_lds(g0, g1, z4, z4, 0);
#endif
}

__device__ __forceinline__ unsigned lds_off_of(const void* p) {
    return (unsigned)(uintptr_t)p;   // LDS aperture: addr[31:0] is the LDS offset
}

// =====================================================================
// f32 -> bf16 elementwise convert
// =====================================================================
__global__ __launch_bounds__(256) void to_bf16_kernel(const float* __restrict__ in,
                                                      bf16_t* __restrict__ out, int n) {
    int i = blockIdx.x * blockDim.x + threadIdx.x;
    if (i < n) out[i] = (bf16_t)in[i];
}

// =====================================================================
// W [K][N] f32 -> Wt [N][K] bf16  (so GEMM B-tiles are row-contiguous)
// =====================================================================
__global__ __launch_bounds__(256) void to_bf16_transpose_kernel(const float* __restrict__ in,
                                                                bf16_t* __restrict__ out,
                                                                int K, int N) {
    int i = blockIdx.x * blockDim.x + threadIdx.x;
    if (i >= K * N) return;
    int n = i % N;
    int k = i / N;
    out[(size_t)n * K + k] = (bf16_t)in[i];
}

// =====================================================================
// V: [T][NKV*HD] f32  ->  [NKV][HD][T] bf16   (transposed for PV B-frags)
// =====================================================================
__global__ __launch_bounds__(256) void v_transpose_kernel(const float* __restrict__ in,
                                                          bf16_t* __restrict__ out) {
    int i = blockIdx.x * blockDim.x + threadIdx.x;
    if (i >= T_SEQ * NKV * HD) return;
    int d = i % HD;
    int h = (i / HD) % NKV;
    int t = i / (HD * NKV);
    out[((size_t)h * HD + d) * T_SEQ + t] = (bf16_t)in[i];
}

// =====================================================================
// bf16 GEMM: C[M,N] = A[M,K] * Bt[N,K]^T, fp32 accumulate.
// TDM-staged, double-buffered LDS. 128 thr = 4 waves; 64x64 block tile.
// =====================================================================
__global__ __launch_bounds__(128) void gemm_bf16_kernel(const bf16_t* __restrict__ A,
                                                        const bf16_t* __restrict__ Bt,
                                                        float* __restrict__ C,
                                                        int M, int N, int K) {
    __shared__ __align__(16) bf16_t As[2][64][32];   // [m][k]
    __shared__ __align__(16) bf16_t Bs[2][64][32];   // [n][k]

    const int tid  = threadIdx.x;
    const int wave = tid >> 5;
    const int lane = tid & 31;
    const int half = lane >> 4;
    const int l15  = lane & 15;
    const int wm   = (wave >> 1) * 32;
    const int wn   = (wave & 1) * 32;
    const int bm   = blockIdx.y * 64;
    const int bn   = blockIdx.x * 64;

    v8f acc[2][2] = {};
    const int nk = K / 32;

    if (wave == 0) {   // prologue DMA: tile 0
        tdm_load_2d(lds_off_of(&As[0][0][0]), A  + (size_t)bm * K, 32, 64, (unsigned long long)K);
        tdm_load_2d(lds_off_of(&Bs[0][0][0]), Bt + (size_t)bn * K, 32, 64, (unsigned long long)K);
    }

    for (int kb = 0; kb < nk; ++kb) {
        const int buf = kb & 1;
        if (wave == 0) {
            if (kb + 1 < nk) {   // pipeline next tile into the other buffer
                tdm_load_2d(lds_off_of(&As[buf ^ 1][0][0]),
                            A + (size_t)bm * K + (kb + 1) * 32, 32, 64, (unsigned long long)K);
                tdm_load_2d(lds_off_of(&Bs[buf ^ 1][0][0]),
                            Bt + (size_t)bn * K + (kb + 1) * 32, 32, 64, (unsigned long long)K);
                __builtin_amdgcn_s_wait_tensorcnt((short)2);   // oldest pair (this buf) done
            } else {
                __builtin_amdgcn_s_wait_tensorcnt((short)0);
            }
        }
        // light L2 prefetch of the tile after next
        if (kb + 2 < nk) {
            __builtin_prefetch(A + (size_t)(bm + (tid >> 1)) * K + (kb + 2) * 32, 0, 0);
            __builtin_prefetch(Bt + (size_t)(bn + (tid >> 1)) * K + (kb + 2) * 32, 0, 0);
        }
        __syncthreads();

        // ---- fragments (ISA 7.12.2 layouts) ----
        Frag16 af[2], bfr[2];
#pragma unroll
        for (int i = 0; i < 2; ++i) {
            const int row = wm + i * 16 + l15;
            af[i].u[0] = *(const uint4*)&As[buf][row][half * 8];        // K 0-7 / 8-15
            af[i].u[1] = *(const uint4*)&As[buf][row][16 + half * 8];   // K 16-23 / 24-31
            const int col = wn + i * 16 + l15;
            bfr[i].u[0] = *(const uint4*)&Bs[buf][col][half * 8];       // K 0-7 / 8-15
            bfr[i].u[1] = *(const uint4*)&Bs[buf][col][16 + half * 8];  // K 16-23 / 24-31
        }
#pragma unroll
        for (int i = 0; i < 2; ++i)
#pragma unroll
            for (int j = 0; j < 2; ++j)
                acc[i][j] = wmma_bf16(af[i].v, bfr[j].v, acc[i][j]);
        __syncthreads();   // all reads of `buf` done before it is DMA-overwritten
    }

    // ---- C store: lanes 0-15 -> M=r, lanes 16-31 -> M=r+8 ----
#pragma unroll
    for (int i = 0; i < 2; ++i)
#pragma unroll
        for (int j = 0; j < 2; ++j) {
            const int col = bn + wn + j * 16 + l15;
#pragma unroll
            for (int r = 0; r < 8; ++r) {
                const int row = bm + wm + i * 16 + r + half * 8;
                C[(size_t)row * N + col] = acc[i][j][r];
            }
        }
}

// =====================================================================
// RMSNorm + golden-gated RoPE; one wave per (head, token) row of 128.
//  in : [T][NH*HD] f32 ; out : [NH][T][HD] bf16
// =====================================================================
__global__ __launch_bounds__(256) void rmsnorm_rope_kernel(const float* __restrict__ in,
                                                           const float* __restrict__ w,
                                                           const int* __restrict__ row_ids,
                                                           const int* __restrict__ col_ids,
                                                           bf16_t* __restrict__ out, int NH) {
    const int wid  = blockIdx.x * 8 + (threadIdx.x >> 5);
    const int lane = threadIdx.x & 31;
    if (wid >= NH * T_SEQ) return;
    const int h = wid / T_SEQ;
    const int t = wid % T_SEQ;

    const float* base = in + (size_t)t * NH * HD + h * HD;
    float x0 = base[lane];
    float x1 = base[lane + 32];
    float x2 = base[lane + 64];
    float x3 = base[lane + 96];

    float ss = x0 * x0 + x1 * x1 + x2 * x2 + x3 * x3;
#pragma unroll
    for (int off = 1; off < 32; off <<= 1) ss += __shfl_xor(ss, off, 32);
    const float rinv = rsqrtf(ss * (1.0f / HD) + 1e-6f);

    float n0 = x0 * rinv * w[lane];
    float n1 = x1 * rinv * w[lane + 32];
    float n2 = x2 * rinv * w[lane + 64];
    float n3 = x3 * rinv * w[lane + 96];

    const float fr = __powf(10000.0f, -(float)lane * (1.0f / 32.0f));
    const float fc = fr * 1.61803398874989f;
    const float ar = (float)row_ids[t] * fr;
    const float ac = (float)col_ids[t] * fc;
    const float cr = __cosf(ar), sr = __sinf(ar);
    const float cc = __cosf(ac), sc = __sinf(ac);

    bf16_t* o = out + ((size_t)h * T_SEQ + t) * HD;
    o[lane]      = (bf16_t)(n0 * cr - n1 * sr);
    o[lane + 32] = (bf16_t)(n0 * sr + n1 * cr);
    o[lane + 64] = (bf16_t)(n2 * cc - n3 * sc);
    o[lane + 96] = (bf16_t)(n2 * sc + n3 * cc);
}

// =====================================================================
// Flash-style GQA attention. 4 waves/block, all the same head; each wave
// owns 16 query rows. K/V 32-key tiles are TDM-staged into double-buffered
// LDS shared by all 4 waves. Per tile: 8 WMMA QK^T + online softmax +
// LDS P-transpose bounce + 8 WMMA PV.
// q: [NQ][T][HD]  k: [NKV][T][HD]  v: [NKV][HD][T]  mask: [T][T]
// out: [T][NQ*HD] bf16
// =====================================================================
__global__ __launch_bounds__(128) void attn_kernel(const bf16_t* __restrict__ qb,
                                                   const bf16_t* __restrict__ kb,
                                                   const bf16_t* __restrict__ vb,
                                                   const float* __restrict__ mask,
                                                   bf16_t* __restrict__ ob, float scale) {
    __shared__ __align__(16) bf16_t Ks[2][32][128];    // [key][d]  16 KB
    __shared__ __align__(16) bf16_t Vs[2][128][32];    // [d][key]  16 KB
    __shared__ __align__(16) bf16_t pbuf[4][16][32];   // per-wave P bounce, 4 KB

    const int tid  = threadIdx.x;
    const int wave = tid >> 5;
    const int lane = tid & 31;
    const int half = lane >> 4;
    const int l15  = lane & 15;

    const int h   = blockIdx.x >> 5;                       // 32 blocks per head
    const int q0  = ((blockIdx.x & 31) * 4 + wave) * 16;
    const int kvh = h / GROUP;

    // --- Q fragments: 4 K-chunks of 32 ---
    Frag16 qf[4];
    const bf16_t* qrow = qb + ((size_t)h * T_SEQ + q0 + l15) * HD;
#pragma unroll
    for (int kc = 0; kc < 4; ++kc) {
        qf[kc].u[0] = *(const uint4*)(qrow + kc * 32 + half * 8);
        qf[kc].u[1] = *(const uint4*)(qrow + kc * 32 + 16 + half * 8);
    }

    v8f oacc[8] = {};
    float mrow[8], lrow[8];
#pragma unroll
    for (int r = 0; r < 8; ++r) { mrow[r] = -1e30f; lrow[r] = 0.0f; }

    const bf16_t* kbase = kb + (size_t)kvh * T_SEQ * HD;
    const bf16_t* vbase = vb + (size_t)kvh * HD * T_SEQ;

    if (wave == 0) {   // prologue DMA: tile 0
        tdm_load_2d(lds_off_of(&Ks[0][0][0]), kbase, 128, 32, (unsigned long long)HD);
        tdm_load_2d(lds_off_of(&Vs[0][0][0]), vbase, 32, 128, (unsigned long long)T_SEQ);
    }

    const int ntiles = T_SEQ / 32;
    for (int it = 0; it < ntiles; ++it) {
        const int buf = it & 1;
        const int kt  = it * 32;
        if (wave == 0) {
            if (it + 1 < ntiles) {
                tdm_load_2d(lds_off_of(&Ks[buf ^ 1][0][0]),
                            kbase + (size_t)(kt + 32) * HD, 128, 32, (unsigned long long)HD);
                tdm_load_2d(lds_off_of(&Vs[buf ^ 1][0][0]),
                            vbase + (kt + 32), 32, 128, (unsigned long long)T_SEQ);
                __builtin_amdgcn_s_wait_tensorcnt((short)2);
            } else {
                __builtin_amdgcn_s_wait_tensorcnt((short)0);
            }
        }
        __syncthreads();

        // ---- scores: two 16-key column tiles ----
        v8f s0 = {}, s1 = {};
#pragma unroll
        for (int kc = 0; kc < 4; ++kc) {
            Frag16 b0, b1;
            b0.u[0] = *(const uint4*)&Ks[buf][l15][kc * 32 + half * 8];
            b0.u[1] = *(const uint4*)&Ks[buf][l15][kc * 32 + 16 + half * 8];
            b1.u[0] = *(const uint4*)&Ks[buf][16 + l15][kc * 32 + half * 8];
            b1.u[1] = *(const uint4*)&Ks[buf][16 + l15][kc * 32 + 16 + half * 8];
            s0 = wmma_bf16(qf[kc].v, b0.v, s0);
            s1 = wmma_bf16(qf[kc].v, b1.v, s1);
        }

        // ---- scale + mask (C layout: row = r + 8*half, col = l15) ----
#pragma unroll
        for (int r = 0; r < 8; ++r) {
            const float* mr = mask + (size_t)(q0 + r + half * 8) * T_SEQ + kt;
            s0[r] = s0[r] * scale + mr[l15];
            s1[r] = s1[r] * scale + mr[16 + l15];
        }

        // ---- online softmax ----
        float corr[8];
#pragma unroll
        for (int r = 0; r < 8; ++r) {
            float tmax = fmaxf(s0[r], s1[r]);
            tmax = fmaxf(tmax, __shfl_xor(tmax, 1, 32));
            tmax = fmaxf(tmax, __shfl_xor(tmax, 2, 32));
            tmax = fmaxf(tmax, __shfl_xor(tmax, 4, 32));
            tmax = fmaxf(tmax, __shfl_xor(tmax, 8, 32));
            const float mnew = fmaxf(mrow[r], tmax);
            corr[r] = __expf(mrow[r] - mnew);
            const float e0 = __expf(s0[r] - mnew);
            const float e1 = __expf(s1[r] - mnew);
            float rs = e0 + e1;
            rs += __shfl_xor(rs, 1, 32);
            rs += __shfl_xor(rs, 2, 32);
            rs += __shfl_xor(rs, 4, 32);
            rs += __shfl_xor(rs, 8, 32);
            lrow[r] = lrow[r] * corr[r] + rs;
            mrow[r] = mnew;
            s0[r] = e0;
            s1[r] = e1;
        }
#pragma unroll
        for (int c = 0; c < 8; ++c)
#pragma unroll
            for (int r = 0; r < 8; ++r) oacc[c][r] *= corr[r];

        // ---- P: C-layout -> LDS row-major -> A-fragment ----
#pragma unroll
        for (int r = 0; r < 8; ++r) {
            pbuf[wave][r + half * 8][l15]      = (bf16_t)s0[r];
            pbuf[wave][r + half * 8][16 + l15] = (bf16_t)s1[r];
        }
        asm volatile("s_wait_dscnt 0" ::: "memory");
        Frag16 pf;
        pf.u[0] = *(const uint4*)&pbuf[wave][l15][half * 8];
        pf.u[1] = *(const uint4*)&pbuf[wave][l15][16 + half * 8];

        // ---- PV: 8 HD-chunks of 16, K = 32 keys ----
#pragma unroll
        for (int c = 0; c < 8; ++c) {
            Frag16 vf;
            vf.u[0] = *(const uint4*)&Vs[buf][c * 16 + l15][half * 8];
            vf.u[1] = *(const uint4*)&Vs[buf][c * 16 + l15][16 + half * 8];
            oacc[c] = wmma_bf16(pf.v, vf.v, oacc[c]);
        }
        __syncthreads();   // everyone done with `buf` before next DMA overwrites it
    }

    // ---- epilogue: /l, write [T][NQ*HD] bf16 ----
#pragma unroll
    for (int c = 0; c < 8; ++c)
#pragma unroll
        for (int r = 0; r < 8; ++r) {
            const int row = q0 + r + half * 8;
            const float val = oacc[c][r] / lrow[r];
            ob[(size_t)row * (NQ * HD) + h * HD + c * 16 + l15] = (bf16_t)val;
        }
}

// =====================================================================
// host launcher
// =====================================================================
extern "C" void kernel_launch(void* const* d_in, const int* in_sizes, int n_in,
                              void* d_out, int out_size, void* d_ws, size_t ws_size,
                              hipStream_t stream) {
    (void)in_sizes; (void)n_in; (void)out_size; (void)ws_size;

    const float* x       = (const float*)d_in[0];
    const int*   row_ids = (const int*)d_in[1];
    const int*   col_ids = (const int*)d_in[2];
    const float* mask    = (const float*)d_in[3];
    const float* Wq      = (const float*)d_in[4];
    const float* Wk      = (const float*)d_in[5];
    const float* Wv      = (const float*)d_in[6];
    const float* Wo      = (const float*)d_in[7];
    const float* qnw     = (const float*)d_in[8];
    const float* knw     = (const float*)d_in[9];
    float* out = (float*)d_out;

    char* ws = (char*)d_ws;
    size_t off = 0;
    auto alloc = [&](size_t bytes) -> void* {
        void* p = ws + off;
        off = (off + bytes + 255) & ~(size_t)255;
        return p;
    };

    const size_t nX  = (size_t)T_SEQ * HID;
    const size_t nWq = (size_t)HID * (NQ * HD);
    const size_t nWk = (size_t)HID * (NKV * HD);

    bf16_t* xb   = (bf16_t*)alloc(nX * 2);
    bf16_t* Wqt  = (bf16_t*)alloc(nWq * 2);                     // [NQ*HD][HID]
    bf16_t* Wkt  = (bf16_t*)alloc(nWk * 2);                     // [NKV*HD][HID]
    bf16_t* Wvt  = (bf16_t*)alloc(nWk * 2);
    bf16_t* Wot  = (bf16_t*)alloc(nWq * 2);                     // [HID][NQ*HD]
    float*  qf   = (float*)alloc(nX * 4);                       // [T][NQ*HD]
    float*  kf   = (float*)alloc((size_t)T_SEQ * NKV * HD * 4);
    float*  vf   = (float*)alloc((size_t)T_SEQ * NKV * HD * 4);
    bf16_t* qbuf = (bf16_t*)alloc(nX * 2);                      // [NQ][T][HD]
    bf16_t* kbuf = (bf16_t*)alloc((size_t)NKV * T_SEQ * HD * 2);
    bf16_t* vbuf = (bf16_t*)alloc((size_t)NKV * HD * T_SEQ * 2);
    bf16_t* ab   = (bf16_t*)alloc(nX * 2);                      // [T][NQ*HD]

    to_bf16_kernel<<<(unsigned)((nX + 255) / 256), 256, 0, stream>>>(x, xb, (int)nX);
    to_bf16_transpose_kernel<<<(unsigned)((nWq + 255) / 256), 256, 0, stream>>>(Wq, Wqt, HID, NQ * HD);
    to_bf16_transpose_kernel<<<(unsigned)((nWk + 255) / 256), 256, 0, stream>>>(Wk, Wkt, HID, NKV * HD);
    to_bf16_transpose_kernel<<<(unsigned)((nWk + 255) / 256), 256, 0, stream>>>(Wv, Wvt, HID, NKV * HD);
    to_bf16_transpose_kernel<<<(unsigned)((nWq + 255) / 256), 256, 0, stream>>>(Wo, Wot, NQ * HD, HID);

    // projections: q/k/v = x @ W  (B passed transposed)
    gemm_bf16_kernel<<<dim3((NQ * HD) / 64, T_SEQ / 64), 128, 0, stream>>>(
        xb, Wqt, qf, T_SEQ, NQ * HD, HID);
    gemm_bf16_kernel<<<dim3((NKV * HD) / 64, T_SEQ / 64), 128, 0, stream>>>(
        xb, Wkt, kf, T_SEQ, NKV * HD, HID);
    gemm_bf16_kernel<<<dim3((NKV * HD) / 64, T_SEQ / 64), 128, 0, stream>>>(
        xb, Wvt, vf, T_SEQ, NKV * HD, HID);

    // norm + rope -> bf16, head-major
    rmsnorm_rope_kernel<<<(NQ * T_SEQ) / 8, 256, 0, stream>>>(qf, qnw, row_ids, col_ids, qbuf, NQ);
    rmsnorm_rope_kernel<<<(NKV * T_SEQ) / 8, 256, 0, stream>>>(kf, knw, row_ids, col_ids, kbuf, NKV);

    // v -> [NKV][HD][T] bf16
    v_transpose_kernel<<<(T_SEQ * NKV * HD) / 256, 256, 0, stream>>>(vf, vbuf);

    // attention
    const float scale = logf((float)T_SEQ) / sqrtf((float)HD);
    attn_kernel<<<(NQ * (T_SEQ / 16)) / 4, 128, 0, stream>>>(qbuf, kbuf, vbuf, mask, ab, scale);

    // output projection -> fp32 d_out (Wo transposed)
    gemm_bf16_kernel<<<dim3(HID / 64, T_SEQ / 64), 128, 0, stream>>>(
        ab, Wot, out, T_SEQ, HID, NQ * HD);
}
